// XtrPairwiseCELoss_73650099192571
// MI455X (gfx1250) — matrix operand order repaired and checked by reading, not compiled, not run on hardware
//
#include <hip/hip_runtime.h>
#include <hip/hip_bf16.h>
#include <math.h>

// ---------------------------------------------------------------------------
// XTR pairwise CE loss for B=64, N=128, S=2048, D=128, TOP_K=128 on gfx1250.
// Two WMMA passes (score GEMM recomputed; inputs live in 192MB L2) + radix
// histogram selection of the 128th-largest score per (b, doc-set).
// ---------------------------------------------------------------------------

#define XB 64
#define XN 128
#define XS 2048
#define XD 128
#define XK 128          // TOP_K
#define SC 128          // doc columns per workgroup chunk
#define NCHUNK (XS / SC)
#define PADH 8
#define LDS_STRIDE (XD + PADH)   // halfs per doc row in LDS (even)
#define NBIN 4096

// workspace layout in 32-bit words
#define OFF_ROWMAX 0u                               // [B][2][N] keys
#define SZ_ROWMAX  (XB * 2 * XN)
#define OFF_CH     (OFF_ROWMAX + SZ_ROWMAX)         // coarse hist [B*2][4096]
#define SZ_CH      (XB * 2 * NBIN)
#define OFF_FH     (OFF_CH + SZ_CH)                 // fine hist   [B*2][4096]
#define SZ_FH      (XB * 2 * NBIN)
#define OFF_SEL    (OFF_FH + SZ_FH)                 // [B*2] x {coarseBin, remaining}
#define SZ_SEL     (XB * 2 * 2)
#define OFF_TKEY   (OFF_SEL + SZ_SEL)               // [B*2] threshold key
#define SZ_TKEY    (XB * 2)
#define WS_WORDS   (OFF_TKEY + SZ_TKEY)

typedef __attribute__((ext_vector_type(16))) _Float16 v16h;
typedef __attribute__((ext_vector_type(8)))  float    v8f;

__device__ __forceinline__ unsigned fkey(float f) {
  unsigned b = __float_as_uint(f);
  return (b & 0x80000000u) ? ~b : (b | 0x80000000u);
}
__device__ __forceinline__ float keyToFloat(unsigned k) {
  unsigned b = (k & 0x80000000u) ? (k ^ 0x80000000u) : ~k;
  return __uint_as_float(b);
}

__global__ void xtr_zero_ws(unsigned* ws, int n) {
  int i = blockIdx.x * 256 + threadIdx.x;
  if (i < n) ws[i] = 0u;
}

// pass 0: coarse histogram (key>>20) + per-row max keys
// pass 1: fine histogram ((key>>8)&0xFFF) restricted to selected coarse bin
__global__ __launch_bounds__(256)
void xtr_score_pass(const float* __restrict__ q,
                    const float* __restrict__ dpos,
                    const float* __restrict__ dneg,
                    unsigned* __restrict__ ws,
                    int pass) {
  __shared__ _Float16 sDoc[SC * LDS_STRIDE];   // ~34.8 KB
  __shared__ unsigned sHist[NBIN];             // 16 KB
  __shared__ unsigned sRowMax[XN];             // 0.5 KB

  const int chunk = blockIdx.x;
  const int set   = blockIdx.y;
  const int b     = blockIdx.z;
  const int tid   = threadIdx.x;
  const int wave  = tid >> 5;
  const int lane  = tid & 31;
  const int m     = lane & 15;    // row / column within 16-wide fragment
  const int hi    = lane >> 4;
  const int rowBase = wave * 16;  // 8 waves * 16 rows = 128 = N

  const float* doc = set ? dneg : dpos;
  const int sBase  = chunk * SC;

  // ---- stage doc chunk into LDS as f16 ----
  for (int i = tid; i < SC * XD; i += 256) {
    int sl = i >> 7, k = i & (XD - 1);
    float v = doc[((size_t)b * XS + sBase + sl) * XD + k];
    sDoc[sl * LDS_STRIDE + k] = (_Float16)v;
  }
  for (int i = tid; i < NBIN; i += 256) sHist[i] = 0u;
  if (tid < XN) sRowMax[tid] = 0u;

  int selBin = -1;
  if (pass == 1) selBin = (int)ws[OFF_SEL + 2 * (b * 2 + set)];
  __syncthreads();

  // ---- A fragments: q rows [rowBase, rowBase+16), K = 0..127 ----
  v16h afrag[4];
#pragma unroll
  for (int kb = 0; kb < 4; ++kb) {
#pragma unroll
    for (int e = 0; e < 16; ++e) {
      const int r = e >> 1, c2 = e & 1;
      const int K = kb * 32 +
                    ((r < 4) ? (2 * r + c2 + 8 * hi)
                             : (16 + 2 * (r - 4) + c2 + 8 * hi));
      afrag[kb][e] =
          (_Float16)q[((size_t)b * XN + rowBase + m) * XD + K];
    }
  }

  // ---- column tiles of 16 over the SC-wide chunk ----
  for (int ct = 0; ct < SC / 16; ++ct) {
    const int col = ct * 16 + m;   // this lane's doc column within chunk
    v8f cacc = {};
#pragma unroll
    for (int kb = 0; kb < 4; ++kb) {
      v16h bfrag;
#pragma unroll
      for (int e = 0; e < 16; ++e) {
        const int K = kb * 32 + ((e < 8) ? (e + 8 * hi) : (16 + 8 * hi + (e - 8)));   // B layout: 32x16, col = lane%16
        bfrag[e] = sDoc[col * LDS_STRIDE + K];
      }
      cacc = __builtin_amdgcn_wmma_f32_16x16x32_f16(
          false, afrag[kb], false, bfrag, (short)0, cacc, false, false);
    }
    // C/D layout: lane n = lane%16, VGPR r -> row = r + 8*hi
#pragma unroll
    for (int r = 0; r < 8; ++r) {
      const float score = cacc[r];
      const unsigned key = fkey(score);
      if (pass == 0) {
        atomicMax(&sRowMax[rowBase + r + 8 * hi], key);
        atomicAdd(&sHist[key >> 20], 1u);
      } else {
        if ((int)(key >> 20) == selBin)
          atomicAdd(&sHist[(key >> 8) & 0xFFFu], 1u);
      }
    }
  }
  __syncthreads();

  // ---- flush LDS accumulators to global workspace ----
  const unsigned base = (unsigned)(b * 2 + set);
  if (pass == 0) {
    for (int i = tid; i < NBIN; i += 256) {
      unsigned v = sHist[i];
      if (v) atomicAdd(&ws[OFF_CH + base * NBIN + i], v);
    }
    if (tid < XN)
      atomicMax(&ws[OFF_ROWMAX + base * XN + tid], sRowMax[tid]);
  } else {
    for (int i = tid; i < NBIN; i += 256) {
      unsigned v = sHist[i];
      if (v) atomicAdd(&ws[OFF_FH + base * NBIN + i], v);
    }
  }
}

__global__ void xtr_select_coarse(unsigned* ws) {
  int i = threadIdx.x;              // 0..127 = b*2+set
  if (i >= XB * 2) return;
  const unsigned* h = ws + OFF_CH + (unsigned)i * NBIN;
  unsigned acc = 0;
  int bin = NBIN - 1;
  for (; bin > 0; --bin) {
    acc += h[bin];
    if (acc >= XK) break;
  }
  unsigned above = acc - h[bin];     // strictly above selected bin
  ws[OFF_SEL + 2 * i]     = (unsigned)bin;
  ws[OFF_SEL + 2 * i + 1] = XK > above ? (XK - above) : 1u;
}

__global__ void xtr_select_fine(unsigned* ws) {
  int i = threadIdx.x;
  if (i >= XB * 2) return;
  const unsigned* h = ws + OFF_FH + (unsigned)i * NBIN;
  const unsigned need = ws[OFF_SEL + 2 * i + 1];
  unsigned acc = 0;
  int bin = NBIN - 1;
  for (; bin > 0; --bin) {
    acc += h[bin];
    if (acc >= need) break;
  }
  const unsigned coarse = ws[OFF_SEL + 2 * i];
  ws[OFF_TKEY + i] = (coarse << 20) | ((unsigned)bin << 8);
}

__global__ void xtr_finalize(const unsigned* __restrict__ ws,
                             float* __restrict__ out) {
  __shared__ float red[XB];
  const int b = threadIdx.x;
  float sp = 0.f;
  if (b < XB) {
    float sc[2];
#pragma unroll
    for (int set = 0; set < 2; ++set) {
      const unsigned base = (unsigned)(b * 2 + set);
      const unsigned T = ws[OFF_TKEY + base];
      const unsigned* rm = ws + OFF_ROWMAX + base * XN;
      float sum = 0.f;
      int cnt = 0;
      for (int n = 0; n < XN; ++n) {
        unsigned k = rm[n];
        if (k >= T) { sum += keyToFloat(k); ++cnt; }
      }
      float Z = fmaxf((float)cnt, 0.001f);
      sc[set] = sum / Z;
    }
    const float x = sc[1] - sc[0];              // neg - pos
    sp = fmaxf(x, 0.f) + log1pf(expf(-fabsf(x)));  // stable softplus
  }
  red[threadIdx.x] = sp;
  __syncthreads();
  if (threadIdx.x == 0) {
    float t = 0.f;
    for (int i = 0; i < XB; ++i) t += red[i];
    out[0] = t / (float)XB;
  }
}

extern "C" void kernel_launch(void* const* d_in, const int* in_sizes, int n_in,
                              void* d_out, int out_size, void* d_ws, size_t ws_size,
                              hipStream_t stream) {
  const float* q    = (const float*)d_in[0];
  const float* dpos = (const float*)d_in[1];
  const float* dneg = (const float*)d_in[2];
  unsigned* ws = (unsigned*)d_ws;
  float* out = (float*)d_out;

  // deterministic per call: re-zero selection state every launch
  xtr_zero_ws<<<(WS_WORDS + 255) / 256, 256, 0, stream>>>(ws, (int)WS_WORDS);

  dim3 grid(NCHUNK, 2, XB);
  xtr_score_pass<<<grid, 256, 0, stream>>>(q, dpos, dneg, ws, 0);
  xtr_select_coarse<<<1, 128, 0, stream>>>(ws);
  xtr_score_pass<<<grid, 256, 0, stream>>>(q, dpos, dneg, ws, 1);
  xtr_select_fine<<<1, 128, 0, stream>>>(ws);
  xtr_finalize<<<1, XB, 0, stream>>>(ws, out);
}
